// SimpleRNN_61933428416961
// MI455X (gfx1250) — compile-verified, matrix operand order checked by reading, not run on hardware
//
#include <hip/hip_runtime.h>
#include <math.h>

// ---------------------------------------------------------------------------
// 2-layer tanh RNN, B=32 T=512 I=512 H=1024, bf16 WMMA w/ f32 accumulate.
//   phase 1: XW = X @ W_ih^T + (b_ih + b_hh)      (time-parallel GEMM,
//            XW stored in WMMA-fragment-major layout)
//   phase 2: h_t = tanh(xw_t + h_{t-1} @ W_hh^T)  (persistent scan kernel,
//            W_hh resident in 128KB LDS per WGP, grid-sync per timestep)
// ---------------------------------------------------------------------------

typedef __bf16 bf16;
typedef __attribute__((ext_vector_type(8)))  bf16  v8bf;
typedef __attribute__((ext_vector_type(16))) bf16  v16bf;
typedef __attribute__((ext_vector_type(8)))  float v8f;
typedef __attribute__((ext_vector_type(4)))  float v4f;

#define RNN_B 32
#define RNN_T 512
#define RNN_I 512
#define RNN_H 1024

// d_ws layout (requires 128 MiB + 256 B of scratch)
#define XW_OFF   0ull              // 64 MiB : f32  fragment-major [mtile][ntile][lane][8]
#define H1_OFF   (64ull << 20)     // 32 MiB : bf16 [T*B][H]   layer-0 h history
#define H2_OFF   (96ull << 20)     // 32 MiB : bf16 [T*B][H]   layer-1 h history
#define SYNC_OFF (128ull << 20)    // sync counters

// --- Fragment loaders matching the CDNA5 16-bit A/B VGPR layout ------------
// Per lane: row fixed (= lane&15 within the tile), g = lane<16 ? 0 : 8.
// The 16 bf16 values of the fragment are two contiguous runs of 8 elements
// at K = kofs and K = kofs+16  (kofs = kblock*32 + g).

static __device__ inline v16bf frag_from_f32(const float* __restrict__ row, int kofs) {
  v4f a = *(const v4f*)(row + kofs);
  v4f b = *(const v4f*)(row + kofs + 4);
  v4f c = *(const v4f*)(row + kofs + 16);
  v4f d = *(const v4f*)(row + kofs + 20);
  v16bf f;
#pragma unroll
  for (int i = 0; i < 4; ++i) {
    f[i]      = (bf16)a[i];
    f[4 + i]  = (bf16)b[i];
    f[8 + i]  = (bf16)c[i];
    f[12 + i] = (bf16)d[i];
  }
  return f;
}

static __device__ inline v16bf frag_from_bf16(const bf16* __restrict__ row, int kofs) {
  v8bf r0 = *(const v8bf*)(row + kofs);        // 16-byte load
  v8bf r1 = *(const v8bf*)(row + kofs + 16);   // 16-byte load
  v16bf f;
#pragma unroll
  for (int i = 0; i < 8; ++i) { f[i] = r0[i]; f[8 + i] = r1[i]; }
  return f;
}

// Branch-free tanh: t = exp(-2|x|); tanh = copysign((1-t)/(1+t), x).
// Lowers to v_exp_f32 + v_rcp_f32 + a few VALU, no EXEC divergence.
static __device__ inline float fast_tanh(float x) {
  const float ax = fabsf(x);
  const float e  = __expf(-2.0f * ax);                       // native v_exp_f32
  const float r  = (1.0f - e) * __builtin_amdgcn_rcpf(1.0f + e);
  return copysignf(r, x);
}

// --- Phase 1: XW = A @ W^T + (b_ih + b_hh) ---------------------------------
// One wave per 16x16 output tile.  M = T*B = 16384 rows, N = H = 1024 cols.
// Output is stored fragment-major: XW[((mt*64 + nt)*32 + lane)*8 + r], which
// is exactly the v8f accumulator — the scan reads it back with 2 b128 loads.
// A_IS_X: A is f32 x[B][T][I] (row m -> x[m%32][m/32][:]); else bf16 [M][K].
template <bool A_IS_X>
__global__ void __launch_bounds__(256)
rnn_gemm_xw(const void* __restrict__ Av,
            const float* __restrict__ W,     // [H][K] row-major f32 (W_ih)
            const float* __restrict__ bih,
            const float* __restrict__ bhh,
            float* __restrict__ XW,          // fragment-major f32
            int K) {
  const int lane = threadIdx.x & 31;
  const int wave = blockIdx.x * (blockDim.x >> 5) + (threadIdx.x >> 5);
  const int NT   = RNN_H / 16;               // 64 column tiles
  const int mt   = wave / NT;                // 0..1023   (== t*2 + batch-tile)
  const int nt   = wave % NT;                // 0..63
  const int l15  = lane & 15;
  const int g    = (lane < 16) ? 0 : 8;

  const int n = nt * 16 + l15;                         // output column (per lane)
  const float* wrow = W + (size_t)n * K;               // B-matrix source: W row n

  const int Mrow = mt * 16 + l15;                      // A-matrix row (per lane)
  const float* xrow = nullptr;
  const bf16*  hrow = nullptr;
  if constexpr (A_IS_X) {
    const int t = Mrow >> 5, b = Mrow & 31;            // XW row m = t*32 + b
    xrow = (const float*)Av + ((size_t)b * RNN_T + t) * RNN_I;
  } else {
    hrow = (const bf16*)Av + (size_t)Mrow * K;
  }

  const float bias = bih[n] + bhh[n];
  v8f c;
#pragma unroll
  for (int i = 0; i < 8; ++i) c[i] = bias;

  for (int kb = 0; kb < K; kb += 32) {
    v16bf a;
    if constexpr (A_IS_X) a = frag_from_f32(xrow, kb + g);
    else                  a = frag_from_bf16(hrow, kb + g);
    v16bf b = frag_from_f32(wrow, kb + g);
    c = __builtin_amdgcn_wmma_f32_16x16x32_bf16(false, a, false, b,
                                                (short)0, c, false, false);
  }

  // Fragment-major store: lane-contiguous, two b128 per lane (1 KB per wave).
  float* dst = XW + (((size_t)mt * 64 + nt) * 32 + lane) * 8;
  *(v4f*)(dst)     = (v4f){c[0], c[1], c[2], c[3]};
  *(v4f*)(dst + 4) = (v4f){c[4], c[5], c[6], c[7]};
}

// --- Phase 2: persistent recurrent scan ------------------------------------
// 16 blocks x 256 threads = 128 waves = the 2 (M) x 64 (N) tile grid of the
// [32 x 1024] recurrent GEMM.  Each block owns 64 output columns and keeps
// the matching 64 rows of W_hh in LDS as bf16 (64*1024*2 = 128 KB).
template <bool HAS_OUT>
__global__ void __launch_bounds__(256, 1)
rnn_scan(const float* __restrict__ XW,     // fragment-major f32
         const float* __restrict__ Whh,    // [H][H] f32
         bf16* __restrict__ hist,          // [T][B][H] bf16 (h history)
         float* __restrict__ out,          // if HAS_OUT: [B][T][H] f32
         unsigned* __restrict__ sync) {
  extern __shared__ bf16 sW[];             // 64 * RNN_H bf16 = 128 KiB

  const int tid  = threadIdx.x;
  const int lane = tid & 31;
  const int wv   = tid >> 5;               // 0..7
  const int mt   = wv & 1;                 // batch-row tile (0..1)
  const int ntl  = wv >> 1;                // local column tile (0..3)
  const int l15  = lane & 15;
  const int g    = (lane < 16) ? 0 : 8;

  // Stage this block's 64 W_hh rows into LDS (fp32 -> bf16).
  {
    const size_t rowbase = (size_t)blockIdx.x * 64;
    for (int idx = tid; idx < 64 * RNN_H; idx += 256) {
      const int r = idx >> 10, col = idx & (RNN_H - 1);
      sW[idx] = (bf16)Whh[(rowbase + r) * RNN_H + col];
    }
  }
  __syncthreads();

  const int   ntG   = blockIdx.x * 4 + ntl;                  // global n tile
  const int   n     = ntG * 16 + l15;                        // global column
  const bf16* swrow = sW + (size_t)(ntl * 16 + l15) * RNN_H; // B frag source row
  const int   arow  = mt * 16 + l15;                         // A frag batch row
  const int   m0    = mt * 16 + ((lane < 16) ? 0 : 8);       // C tile rows

  for (int t = 0; t < RNN_T; ++t) {
    // xw tile in fragment-major layout: two coalesced b128 loads per lane.
    const float* src = XW + (((size_t)(t * 2 + mt) * 64 + ntG) * 32 + lane) * 8;
    const v4f lo = *(const v4f*)(src);
    const v4f hi = *(const v4f*)(src + 4);
    v8f c;
#pragma unroll
    for (int i = 0; i < 4; ++i) { c[i] = lo[i]; c[4 + i] = hi[i]; }

    if (t + 1 < RNN_T)  // prefetch next step's xw fragment
      __builtin_prefetch(src + (size_t)2 * 64 * 32 * 8, 0, 1);

    if (t > 0) {
      const bf16* hprev =
          hist + (size_t)(t - 1) * (RNN_B * RNN_H) + (size_t)arow * RNN_H;
#pragma unroll 4
      for (int kb = 0; kb < RNN_H; kb += 32) {
        v16bf a = frag_from_bf16(hprev, kb + g);
        v16bf b = frag_from_bf16(swrow, kb + g);
        c = __builtin_amdgcn_wmma_f32_16x16x32_bf16(false, a, false, b,
                                                    (short)0, c, false, false);
      }
    }

    // 8 independent tanh chains first (lets the compiler pipeline the
    // exp->rcp->mul dependency chains across elements), then the stores.
    float v[8];
#pragma unroll
    for (int r = 0; r < 8; ++r) v[r] = fast_tanh(c[r]);

    bf16* hcur = hist + (size_t)t * (RNN_B * RNN_H);
#pragma unroll
    for (int r = 0; r < 8; ++r)
      hcur[(size_t)(m0 + r) * RNN_H + n] = (bf16)v[r];       // recurrence state (RT)

    if constexpr (HAS_OUT) {
#pragma unroll
      for (int r = 0; r < 8; ++r)                            // final output (NT)
        __builtin_nontemporal_store(v[r], out + ((size_t)(m0 + r) * RNN_T + t) * RNN_H + n);
    }

    // Device-wide barrier between timesteps (release stores, then arrive+spin).
    __threadfence();
    __syncthreads();
    if (tid == 0) {
      atomicAdd(sync, 1u);
      const unsigned target = (unsigned)gridDim.x * (unsigned)(t + 1);
      while (__hip_atomic_load(sync, __ATOMIC_RELAXED, __HIP_MEMORY_SCOPE_AGENT) < target)
        __builtin_amdgcn_s_sleep(2);
    }
    __syncthreads();
    __threadfence();
  }
}

// ---------------------------------------------------------------------------
extern "C" void kernel_launch(void* const* d_in, const int* in_sizes, int n_in,
                              void* d_out, int out_size, void* d_ws, size_t ws_size,
                              hipStream_t stream) {
  const float* x    = (const float*)d_in[0];
  const float* Wih0 = (const float*)d_in[1];
  const float* Whh0 = (const float*)d_in[2];
  const float* bih0 = (const float*)d_in[3];
  const float* bhh0 = (const float*)d_in[4];
  const float* Wih1 = (const float*)d_in[5];
  const float* Whh1 = (const float*)d_in[6];
  const float* bih1 = (const float*)d_in[7];
  const float* bhh1 = (const float*)d_in[8];

  char*     ws   = (char*)d_ws;
  float*    XW   = (float*)(ws + XW_OFF);
  bf16*     H1   = (bf16*)(ws + H1_OFF);
  bf16*     H2   = (bf16*)(ws + H2_OFF);
  unsigned* sync = (unsigned*)(ws + SYNC_OFF);

  hipMemsetAsync(sync, 0, 256, stream);   // reset grid-sync counters (capturable)

  const int tilesM = (RNN_T * RNN_B) / 16;          // 1024
  const int tilesN = RNN_H / 16;                    // 64
  const int gemmBlocks = (tilesM * tilesN) / 8;     // 8 waves per 256-thread block

  const size_t ldsBytes = (size_t)64 * RNN_H * sizeof(bf16);  // 128 KiB

  // Layer 0
  rnn_gemm_xw<true><<<gemmBlocks, 256, 0, stream>>>(x, Wih0, bih0, bhh0, XW, RNN_I);
  rnn_scan<false><<<16, 256, ldsBytes, stream>>>(XW, Whh0, H1, nullptr, sync + 0);
  // Layer 1
  rnn_gemm_xw<false><<<gemmBlocks, 256, 0, stream>>>(H1, Wih1, bih1, bhh1, XW, RNN_H);
  rnn_scan<true><<<16, 256, ldsBytes, stream>>>(XW, Whh1, H2, (float*)d_out, sync + 16);
}
